// Encoder_54004918780712
// MI455X (gfx1250) — compile-verified, matrix operand order checked by reading, not run on hardware
//
#include <hip/hip_runtime.h>
#include <stdint.h>

// ---------------- problem constants (from reference) ----------------
#define BDIM   512      // model dim D
#define SEQ    2048     // sequence length S
#define BATCH  4
#define HEADS  8
#define DH     64       // head dim
#define FFN    2048
#define LAYERS 6
#define MTOT   (BATCH * SEQ)   // 8192 rows
#define LN_EPS 1e-5f

typedef __attribute__((ext_vector_type(16))) __bf16 bf16x16;
typedef __attribute__((ext_vector_type(8)))  float  f32x8;

union FragBF {
  bf16x16 v;
  uint4   q[2];
};

// round-to-nearest-even f32 -> bf16 (pure integer, no hw cvt dependence)
__device__ __forceinline__ __bf16 f2bf(float f) {
  union { float f; uint32_t u; } c; c.f = f;
  uint32_t r = c.u + 0x7FFFu + ((c.u >> 16) & 1u);
  unsigned short h = (unsigned short)(r >> 16);
  __bf16 o;
  __builtin_memcpy(&o, &h, 2);
  return o;
}

// Load one 16x32 bf16 WMMA fragment (A layout; B uses the mirrored layout
// with the 16-wide dim as "row"). Per ISA 7.12.2:
//   lanes 0-15 : row = lane,    K in {0..7, 16..23}
//   lanes 16-31: row = lane-16, K in {8..15, 24..31}
__device__ __forceinline__ FragBF load_frag(const __bf16* base, int stride) {
  const int lane = threadIdx.x & 31;
  const int r    = lane & 15;
  const int kh   = (lane >> 4) << 3;   // 0 or 8
  const __bf16* p = base + (size_t)r * stride + kh;
  FragBF f;
  f.q[0] = *(const uint4*)(p);         // K = kh .. kh+7
  f.q[1] = *(const uint4*)(p + 16);    // K = kh+16 .. kh+23
  return f;
}

__device__ __forceinline__ f32x8 wmma_bf16(const FragBF& a, const FragBF& b, f32x8 c) {
  return __builtin_amdgcn_wmma_f32_16x16x32_bf16(
      /*neg_a=*/false, a.v, /*neg_b=*/false, b.v,
      /*c_mod=*/(short)0, c, /*reuse_a=*/false, /*reuse_b=*/false);
}

// CDNA5 async memory->LDS copy, 16 bytes per lane, tracked with ASYNCcnt.
__device__ __forceinline__ void async_ld_b128(const void* lds_ptr, const void* gptr) {
  uint32_t l = (uint32_t)(uintptr_t)lds_ptr;     // low 32 bits = LDS byte offset
  uint64_t g = (uint64_t)(uintptr_t)gptr;
  asm volatile("global_load_async_to_lds_b128 %0, %1, off"
               :: "v"(l), "v"(g) : "memory");
}
__device__ __forceinline__ void wait_async0() {
  asm volatile("s_wait_asynccnt 0" ::: "memory");
}

// ---------------- weight convert: fp32 [K,N] -> bf16 [N,K] ----------------
__global__ void wcvt_kernel(const float* __restrict__ W, __bf16* __restrict__ Wt,
                            int K, int N) {
  size_t idx = (size_t)blockIdx.x * blockDim.x + threadIdx.x;  // over N*K
  if (idx >= (size_t)N * K) return;
  int k = (int)(idx % K);
  int n = (int)(idx / K);
  Wt[idx] = f2bf(W[(size_t)k * N + n]);
}

// ---------------- embedding gather ----------------
__global__ void embed_kernel(const int* __restrict__ src, const float* __restrict__ emb,
                             float* __restrict__ xf, __bf16* __restrict__ xb) {
  size_t idx = (size_t)blockIdx.x * blockDim.x + threadIdx.x;  // over M*D
  if (idx >= (size_t)MTOT * BDIM) return;
  int    d   = (int)(idx & (BDIM - 1));
  size_t row = idx >> 9;                   // BDIM = 512
  int    tok = src[row];
  float  v   = emb[(size_t)tok * BDIM + d];
  xf[idx] = v;
  xb[idx] = f2bf(v);
}

// ---------------- V -> V^T [B,H,Dh,S] ----------------
__global__ void vtrans_kernel(const __bf16* __restrict__ V, __bf16* __restrict__ Vt) {
  size_t idx = (size_t)blockIdx.x * blockDim.x + threadIdx.x;  // over M*D
  if (idx >= (size_t)MTOT * BDIM) return;
  int    d  = (int)(idx & (BDIM - 1));
  size_t r  = idx >> 9;
  int    s  = (int)(r & (SEQ - 1));
  int    b  = (int)(r >> 11);              // SEQ = 2048
  int    h  = d >> 6;                      // DH = 64
  int    dh = d & 63;
  int    bh = b * HEADS + h;
  Vt[((size_t)bh * DH + dh) * SEQ + s] = V[idx];
}

// ---------------- tiled WMMA GEMM: out = act(X[M,K] @ Wt[N,K]^T + bias) ----------------
// block = 256 thr (8 wave32), tile 64x128, BK=32, double-buffered async LDS staging.
// M%64==0, N%128==0, K%32==0.
#define GBM 64
#define GBN 128
#define GBK 32
#define LDK 40   // padded LDS row stride (bf16 elems)

__global__ __launch_bounds__(256) void gemm_bf16_kernel(
    const __bf16* __restrict__ X, const __bf16* __restrict__ Wt,
    const float* __restrict__ bias,
    float* __restrict__ outF, __bf16* __restrict__ outB,
    int M, int N, int K, int relu)
{
  __shared__ __attribute__((aligned(16))) __bf16 sX[2][GBM * LDK];
  __shared__ __attribute__((aligned(16))) __bf16 sW[2][GBN * LDK];

  const int tid  = threadIdx.x;
  const int lane = tid & 31;
  const int wid  = tid >> 5;
  const int wm   = wid & 3;        // 16-row strip within tile
  const int wn   = wid >> 2;       // 64-col strip within tile
  const int m0   = blockIdx.y * GBM;
  const int n0   = blockIdx.x * GBN;

  f32x8 acc[4] = {};

  // async-stage one 64x32 X tile + 128x32 W tile into LDS buffer `bufi`
  auto stage = [&](int kt, int bufi) {
    const int k0 = kt * GBK;
    {
      int r = tid >> 2, c = (tid & 3) << 3;            // 256 x b128 for X
      async_ld_b128(&sX[bufi][r * LDK + c],
                    X + (size_t)(m0 + r) * K + k0 + c);
    }
#pragma unroll
    for (int s = 0; s < 2; ++s) {                      // 512 x b128 for W
      int t = tid + s * 256;
      int r = t >> 2, c = (t & 3) << 3;
      async_ld_b128(&sW[bufi][r * LDK + c],
                    Wt + (size_t)(n0 + r) * K + k0 + c);
    }
  };

  const int nK = K / GBK;
  stage(0, 0);
  wait_async0();
  __syncthreads();

  for (int kt = 0; kt < nK; ++kt) {
    const int cur = kt & 1;
    if (kt + 1 < nK) stage(kt + 1, cur ^ 1);   // prefetch next tile (other buffer)

    FragBF a = load_frag(&sX[cur][wm * 16 * LDK], LDK);
#pragma unroll
    for (int j = 0; j < 4; ++j) {
      FragBF b = load_frag(&sW[cur][(wn * 64 + j * 16) * LDK], LDK);
      acc[j] = wmma_bf16(a, b, acc[j]);
    }

    wait_async0();     // my async writes for tile kt+1 are done
    __syncthreads();   // everyone's are done; safe to consume next iteration
  }

  // epilogue: C layout -> VGPR i: lanes 0-15 (M=i, N=lane), lanes 16-31 (M=8+i)
  const int half = lane >> 4;
  const int colL = lane & 15;
#pragma unroll
  for (int j = 0; j < 4; ++j) {
    int   gc = n0 + wn * 64 + j * 16 + colL;
    float bv = bias ? bias[gc] : 0.0f;
#pragma unroll
    for (int i = 0; i < 8; ++i) {
      int   gr = m0 + wm * 16 + half * 8 + i;
      float v  = acc[j][i] + bv;
      if (relu) v = fmaxf(v, 0.0f);
      if (outF) outF[(size_t)gr * N + gc] = v;
      if (outB) outB[(size_t)gr * N + gc] = f2bf(v);
    }
  }
}

// ---------------- flash attention (causal), one 16-row Q tile per wave ----------------
#define PSTR 40
__global__ __launch_bounds__(128) void attn_kernel(
    const __bf16* __restrict__ Q, const __bf16* __restrict__ Kb,
    const __bf16* __restrict__ Vt, __bf16* __restrict__ O)
{
  __shared__ __attribute__((aligned(16))) __bf16 sP[4][16 * PSTR];
  const int tid  = threadIdx.x;
  const int lane = tid & 31;
  const int wid  = tid >> 5;
  const int task = blockIdx.x * 4 + wid;          // B*H*(S/16) = 4096 tasks
  const int qt   = task & 127;
  const int bh   = task >> 7;
  const int b    = bh >> 3;
  const int h    = bh & 7;
  const int q0   = qt * 16;

  const __bf16* Qb = Q  + ((size_t)b * SEQ) * BDIM + h * DH;
  const __bf16* Kp = Kb + ((size_t)b * SEQ) * BDIM + h * DH;
  const __bf16* Vb = Vt + ((size_t)bh * DH) * SEQ;

  FragBF qa0 = load_frag(Qb + (size_t)q0 * BDIM,      BDIM);
  FragBF qa1 = load_frag(Qb + (size_t)q0 * BDIM + 32, BDIM);

  f32x8 o[4] = {};
  float mi[8], li[8];
#pragma unroll
  for (int i = 0; i < 8; ++i) { mi[i] = -1e30f; li[i] = 0.0f; }

  const int   half = lane >> 4;
  const int   colL = lane & 15;
  const float sc   = 0.125f;   // 1/sqrt(Dh)

  for (int n0 = 0; n0 < q0 + 16; n0 += 32) {
    // scores for 32 keys: two 16x16 tiles, Dh=64 -> 2 WMMAs each
    f32x8 s0 = {}, s1 = {};
    {
      FragBF k00 = load_frag(Kp + (size_t)n0 * BDIM,             BDIM);
      FragBF k01 = load_frag(Kp + (size_t)n0 * BDIM + 32,        BDIM);
      s0 = wmma_bf16(qa0, k00, s0);
      s0 = wmma_bf16(qa1, k01, s0);
      FragBF k10 = load_frag(Kp + (size_t)(n0 + 16) * BDIM,      BDIM);
      FragBF k11 = load_frag(Kp + (size_t)(n0 + 16) * BDIM + 32, BDIM);
      s1 = wmma_bf16(qa0, k10, s1);
      s1 = wmma_bf16(qa1, k11, s1);
    }

    __bf16* pbuf = &sP[wid][0];
#pragma unroll
    for (int i = 0; i < 8; ++i) {
      int   qrow = q0 + half * 8 + i;
      float x0 = s0[i] * sc;
      float x1 = s1[i] * sc;
      if (n0 + colL > qrow)      x0 = -1e30f;   // causal mask
      if (n0 + 16 + colL > qrow) x1 = -1e30f;
      float mx = fmaxf(x0, x1);
#pragma unroll
      for (int off = 1; off < 16; off <<= 1)
        mx = fmaxf(mx, __shfl_xor(mx, off, 32));
      float mnew = fmaxf(mi[i], mx);
      float p0 = __expf(x0 - mnew);
      float p1 = __expf(x1 - mnew);
      float rs = p0 + p1;
#pragma unroll
      for (int off = 1; off < 16; off <<= 1)
        rs += __shfl_xor(rs, off, 32);
      float corr = __expf(mi[i] - mnew);
      mi[i] = mnew;
      li[i] = li[i] * corr + rs;
#pragma unroll
      for (int j = 0; j < 4; ++j) o[j][i] *= corr;
      // stage P (C layout) into LDS so we can re-read it in A-fragment layout
      pbuf[(half * 8 + i) * PSTR + colL]      = f2bf(p0);
      pbuf[(half * 8 + i) * PSTR + colL + 16] = f2bf(p1);
    }
    asm volatile("s_wait_dscnt 0" ::: "memory");   // same-wave LDS RAW fence

    FragBF pf = load_frag(pbuf, PSTR);             // P as 16x32 A fragment
#pragma unroll
    for (int j = 0; j < 4; ++j) {
      FragBF vf = load_frag(Vb + (size_t)(j * 16) * SEQ + n0, SEQ);
      o[j] = wmma_bf16(pf, vf, o[j]);
    }
  }

  __bf16* Ob = O + ((size_t)b * SEQ) * BDIM + h * DH;
#pragma unroll
  for (int i = 0; i < 8; ++i) {
    float inv = 1.0f / li[i];
    int   gr  = q0 + half * 8 + i;
#pragma unroll
    for (int j = 0; j < 4; ++j)
      Ob[(size_t)gr * BDIM + j * 16 + colL] = f2bf(o[j][i] * inv);
  }
}

// ---------------- residual add + layernorm (D=512, block=256) ----------------
__global__ __launch_bounds__(256) void add_ln_kernel(
    float* __restrict__ x, const float* __restrict__ y,
    const float* __restrict__ g, const float* __restrict__ bta,
    __bf16* __restrict__ xb)
{
  __shared__ float red[16];
  const int row = blockIdx.x;
  const int tid = threadIdx.x;
  float v[2];
#pragma unroll
  for (int j = 0; j < 2; ++j) {
    size_t o = (size_t)row * BDIM + tid + j * 256;
    v[j] = x[o] + y[o];
  }
  float s  = v[0] + v[1];
  float s2 = v[0] * v[0] + v[1] * v[1];
#pragma unroll
  for (int off = 1; off < 32; off <<= 1) {
    s  += __shfl_xor(s,  off, 32);
    s2 += __shfl_xor(s2, off, 32);
  }
  if ((tid & 31) == 0) { red[tid >> 5] = s; red[8 + (tid >> 5)] = s2; }
  __syncthreads();
  float ts = 0.0f, ts2 = 0.0f;
#pragma unroll
  for (int w = 0; w < 8; ++w) { ts += red[w]; ts2 += red[8 + w]; }
  float mean = ts / (float)BDIM;
  float var  = ts2 / (float)BDIM - mean * mean;
  float rstd = rsqrtf(var + LN_EPS);
#pragma unroll
  for (int j = 0; j < 2; ++j) {
    int    d  = tid + j * 256;
    size_t o  = (size_t)row * BDIM + d;
    float  nv = (v[j] - mean) * rstd * g[d] + bta[d];
    x[o]  = nv;
    xb[o] = f2bf(nv);
  }
}

// ---------------- host orchestration ----------------
extern "C" void kernel_launch(void* const* d_in, const int* in_sizes, int n_in,
                              void* d_out, int out_size, void* d_ws, size_t ws_size,
                              hipStream_t stream) {
  (void)in_sizes; (void)n_in; (void)out_size; (void)ws_size;
  const int*   src  = (const int*)d_in[0];
  const float* emb  = (const float*)d_in[1];
  const float* ln_g = (const float*)d_in[2];
  const float* ln_b = (const float*)d_in[3];
  const float* wq = (const float*)d_in[4];
  const float* bq = (const float*)d_in[5];
  const float* wk = (const float*)d_in[6];
  const float* bk = (const float*)d_in[7];
  const float* wv = (const float*)d_in[8];
  const float* bv = (const float*)d_in[9];
  const float* wo = (const float*)d_in[10];
  const float* bo = (const float*)d_in[11];
  const float* w1 = (const float*)d_in[12];
  const float* b1 = (const float*)d_in[13];
  const float* w2 = (const float*)d_in[14];
  const float* b2 = (const float*)d_in[15];

  char*  ws  = (char*)d_ws;
  size_t off = 0;
  auto alloc = [&](size_t bytes) -> void* {
    off = (off + 255) & ~(size_t)255;
    void* p = ws + off;
    off += bytes;
    return p;
  };

  const size_t MD = (size_t)MTOT * BDIM;   // 8192*512
  const size_t MF = (size_t)MTOT * FFN;    // 8192*2048
  const size_t DD = (size_t)BDIM * BDIM;
  const size_t DF = (size_t)BDIM * FFN;
  const size_t WPL = 4 * DD + 2 * DF;      // bf16 elems per layer

  float*  xf32 = (float*)alloc(MD * 4);
  float*  yf32 = (float*)alloc(MD * 4);
  __bf16* xbf  = (__bf16*)alloc(MD * 2);
  __bf16* qbf  = (__bf16*)alloc(MD * 2);
  __bf16* kbf  = (__bf16*)alloc(MD * 2);
  __bf16* vbf  = (__bf16*)alloc(MD * 2);
  __bf16* att  = (__bf16*)alloc(MD * 2);
  __bf16* vtb  = (__bf16*)alloc(MD * 2);
  __bf16* h1   = (__bf16*)alloc(MF * 2);
  __bf16* wts  = (__bf16*)alloc(WPL * LAYERS * 2);

  // convert + transpose all weights to bf16 [N,K]
  auto cvt = [&](const float* W, __bf16* out, int K, int N) {
    size_t n = (size_t)K * N;
    wcvt_kernel<<<dim3((unsigned)((n + 255) / 256)), dim3(256), 0, stream>>>(W, out, K, N);
  };
  for (int l = 0; l < LAYERS; ++l) {
    __bf16* wl = wts + (size_t)l * WPL;
    cvt(wq + (size_t)l * DD, wl + 0 * DD, BDIM, BDIM);
    cvt(wk + (size_t)l * DD, wl + 1 * DD, BDIM, BDIM);
    cvt(wv + (size_t)l * DD, wl + 2 * DD, BDIM, BDIM);
    cvt(wo + (size_t)l * DD, wl + 3 * DD, BDIM, BDIM);
    cvt(w1 + (size_t)l * DF, wl + 4 * DD, BDIM, FFN);        // K=D, N=F
    cvt(w2 + (size_t)l * DF, wl + 4 * DD + DF, FFN, BDIM);   // K=F, N=D
  }

  embed_kernel<<<dim3((unsigned)((MD + 255) / 256)), 256, 0, stream>>>(src, emb, xf32, xbf);

  auto gemm = [&](const __bf16* X, const __bf16* Wt, const float* bias,
                  float* oF, __bf16* oB, int N, int K, int relu) {
    dim3 grid((unsigned)(N / GBN), (unsigned)(MTOT / GBM));
    gemm_bf16_kernel<<<grid, 256, 0, stream>>>(X, Wt, bias, oF, oB, MTOT, N, K, relu);
  };

  for (int l = 0; l < LAYERS; ++l) {
    __bf16* wl = wts + (size_t)l * WPL;
    gemm(xbf, wl + 0 * DD, bq + (size_t)l * BDIM, nullptr, qbf, BDIM, BDIM, 0);
    gemm(xbf, wl + 1 * DD, bk + (size_t)l * BDIM, nullptr, kbf, BDIM, BDIM, 0);
    gemm(xbf, wl + 2 * DD, bv + (size_t)l * BDIM, nullptr, vbf, BDIM, BDIM, 0);
    vtrans_kernel<<<dim3((unsigned)((MD + 255) / 256)), 256, 0, stream>>>(vbf, vtb);
    attn_kernel<<<dim3(1024), 128, 0, stream>>>(qbf, kbf, vtb, att);
    gemm(att, wl + 3 * DD, bo + (size_t)l * BDIM, yf32, nullptr, BDIM, BDIM, 0);
    add_ln_kernel<<<dim3(MTOT), 256, 0, stream>>>(xf32, yf32, ln_g, ln_b, xbf);
    gemm(xbf, wl + 4 * DD, b1 + (size_t)l * FFN, nullptr, h1, FFN, BDIM, 1);
    gemm(h1, wl + 4 * DD + DF, b2 + (size_t)l * BDIM, yf32, nullptr, BDIM, FFN, 1);
    add_ln_kernel<<<dim3(MTOT), 256, 0, stream>>>(xf32, yf32, ln_g, ln_b, xbf);
  }

  hipMemcpyAsync(d_out, xf32, MD * sizeof(float), hipMemcpyDeviceToDevice, stream);
}
